// RecurrentNodeMemory_76836964926328
// MI455X (gfx1250) — matrix operand order repaired and checked by reading, not compiled, not run on hardware
//
#include <hip/hip_runtime.h>
#include <math.h>

// ---- problem constants (from reference) ----
#define DIMK   128        // hidden dim / GEMM K
#define GATES  384        // 3*DIM
#define WSTR   392        // padded LDS row stride in halves (384+8, breaks bank pattern)
#define MOM    0.9f

typedef __attribute__((ext_vector_type(16))) _Float16 v16h;
typedef __attribute__((ext_vector_type(8)))  float    v8f;

__device__ __forceinline__ float sigm(float x) {
  return 1.0f / (1.0f + __expf(-x));
}

// D = A(16x32 f16) * B(32x16 f16) + C(16x16 f32), wave32
__device__ __forceinline__ v8f wmma_f16(v16h a, v16h b, v8f c) {
  return __builtin_amdgcn_wmma_f32_16x16x32_f16(false, a, false, b, (short)0, c,
                                                false, false);
}

__global__ __launch_bounds__(256)
void gru_scatter_kernel(const float* __restrict__ hidden,
                        const float* __restrict__ variance,
                        const int*   __restrict__ idx,
                        const float* __restrict__ xin,
                        const float* __restrict__ Wih,
                        const float* __restrict__ Whh,
                        const float* __restrict__ bih,
                        const float* __restrict__ bhh,
                        float* __restrict__ out0,
                        float* __restrict__ out1,
                        int batch)
{
  // Stage transposed f16 weights + f32 biases in LDS (shared by 8 waves).
  __shared__ __align__(16) _Float16 sWih[DIMK * WSTR];   // sWih[k*WSTR+n] = Wih[n][k]
  __shared__ __align__(16) _Float16 sWhh[DIMK * WSTR];
  __shared__ float sBih[GATES];
  __shared__ float sBhh[GATES];

  const int tid = threadIdx.x;
  for (int t = tid; t < GATES * DIMK; t += 256) {
    const int n = t / DIMK;       // gate row of W
    const int k = t - n * DIMK;   // input column of W
    sWih[k * WSTR + n] = (_Float16)Wih[t];
    sWhh[k * WSTR + n] = (_Float16)Whh[t];
  }
  for (int t = tid; t < GATES; t += 256) { sBih[t] = bih[t]; sBhh[t] = bhh[t]; }
  __syncthreads();

  const int wave = tid >> 5;
  const int lane = tid & 31;
  const int m0   = blockIdx.x * 128 + wave * 16;   // this wave's 16-row batch tile

  // ---------- A fragments: 16-bit A layout (lane = M, K split by lane-half) ----------
  // lane<16: halves 0..7 -> K = kc*32 + 0..7,  halves 8..15 -> K = kc*32+16 + 0..7
  // lane>=16: same with +8 on the K sub-offset
  const int mrow = lane & 15;
  const int kb   = (lane >> 4) * 8;
  int arow = m0 + mrow; if (arow >= batch) arow = batch - 1;
  const int anode = idx[arow];

  v16h ax[4], ah[4];
  {
    const float* xr = xin    + (size_t)arow  * DIMK;
    const float* hr = hidden + (size_t)anode * DIMK;
#pragma unroll
    for (int kc = 0; kc < 4; ++kc) {
      const int c0 = kc * 32 + kb;
#pragma unroll
      for (int j = 0; j < 8; ++j) {
        ax[kc][j]     = (_Float16)xr[c0 + j];
        ax[kc][8 + j] = (_Float16)xr[c0 + 16 + j];
        ah[kc][j]     = (_Float16)hr[c0 + j];
        ah[kc][8 + j] = (_Float16)hr[c0 + 16 + j];
      }
    }
  }

  // ---------- C/D layout mapping: element (M = v + 8*(lane>=16), N = lane%16) ----------
  const int nl  = lane & 15;
  const int mhi = (lane >> 4) * 8;
  int  cnode[8];
  bool cok[8];
#pragma unroll
  for (int v = 0; v < 8; ++v) {
    const int m = m0 + mhi + v;
    cok[v] = (m < batch);
    cnode[v] = idx[cok[v] ? m : (batch - 1)];
  }

  // ---------- main loop over 8 output column tiles ----------
  for (int dt = 0; dt < 8; ++dt) {
    const int nr = dt * 16;             // reset-gate columns
    const int nz = DIMK + dt * 16;      // update-gate columns
    const int nn = 2 * DIMK + dt * 16;  // new-gate columns

    v8f gi_r = {}, gi_z = {}, gi_n = {}, gh_r = {}, gh_z = {}, gh_n = {};

#pragma unroll
    for (int kc = 0; kc < 4; ++kc) {
      // B layout: lane = N column (n0+lane%16), halves 0..7 -> K = kc*32+kb+0..7, halves 8..15 -> K = kc*32+16+kb+0..7
      const _Float16* bi = &sWih[(size_t)(kc * 32 + kb) * WSTR + nl];
      const _Float16* bh = &sWhh[(size_t)(kc * 32 + kb) * WSTR + nl];
      v16h b;
#pragma unroll
      for (int j = 0; j < 16; ++j) b[j] = bi[(j + (j & 8)) * WSTR + nr];
      gi_r = wmma_f16(ax[kc], b, gi_r);
#pragma unroll
      for (int j = 0; j < 16; ++j) b[j] = bi[(j + (j & 8)) * WSTR + nz];
      gi_z = wmma_f16(ax[kc], b, gi_z);
#pragma unroll
      for (int j = 0; j < 16; ++j) b[j] = bi[(j + (j & 8)) * WSTR + nn];
      gi_n = wmma_f16(ax[kc], b, gi_n);
#pragma unroll
      for (int j = 0; j < 16; ++j) b[j] = bh[(j + (j & 8)) * WSTR + nr];
      gh_r = wmma_f16(ah[kc], b, gh_r);
#pragma unroll
      for (int j = 0; j < 16; ++j) b[j] = bh[(j + (j & 8)) * WSTR + nz];
      gh_z = wmma_f16(ah[kc], b, gh_z);
#pragma unroll
      for (int j = 0; j < 16; ++j) b[j] = bh[(j + (j & 8)) * WSTR + nn];
      gh_n = wmma_f16(ah[kc], b, gh_n);
    }

    // per-lane biases (column N = nl is fixed per lane for all 8 rows)
    const float b_r  = sBih[nr + nl] + sBhh[nr + nl];
    const float b_z  = sBih[nz + nl] + sBhh[nz + nl];
    const float bi_n = sBih[nn + nl];
    const float bh_n = sBhh[nn + nl];
    const int   col  = dt * 16 + nl;

#pragma unroll
    for (int v = 0; v < 8; ++v) {
      const float r  = sigm(gi_r[v] + gh_r[v] + b_r);
      const float z  = sigm(gi_z[v] + gh_z[v] + b_z);
      const float nv = tanhf(gi_n[v] + bi_n + r * (gh_n[v] + bh_n));
      const size_t off = (size_t)cnode[v] * DIMK + col;
      const float hp = hidden[off];
      const float hn = (1.0f - z) * nv + z * hp;
      const float d  = hn - hp;
      const float vn = MOM * variance[off] + (1.0f - MOM) * d * d;
      if (cok[v]) { out0[off] = hn; out1[off] = vn; }
    }
  }
}

extern "C" void kernel_launch(void* const* d_in, const int* in_sizes, int n_in,
                              void* d_out, int out_size, void* d_ws, size_t ws_size,
                              hipStream_t stream) {
  const float* hidden   = (const float*)d_in[0];
  const float* variance = (const float*)d_in[1];
  const int*   idx      = (const int*)  d_in[2];
  const float* new_repr = (const float*)d_in[3];
  const float* W_ih     = (const float*)d_in[4];
  const float* W_hh     = (const float*)d_in[5];
  const float* b_ih     = (const float*)d_in[6];
  const float* b_hh     = (const float*)d_in[7];

  const size_t half   = (size_t)out_size / 2;      // 500000*128 elements
  float* out0 = (float*)d_out;
  float* out1 = (float*)d_out + half;

  // Bulk copy: out[0] = hidden, out[1] = variance (D2D, graph-capturable)
  hipMemcpyAsync(out0, hidden,   half * sizeof(float), hipMemcpyDeviceToDevice, stream);
  hipMemcpyAsync(out1, variance, half * sizeof(float), hipMemcpyDeviceToDevice, stream);

  const int batch  = in_sizes[2];                  // 131072
  const int blocks = (batch + 127) / 128;          // 128 batch rows per block
  gru_scatter_kernel<<<blocks, 256, 0, stream>>>(
      hidden, variance, idx, new_repr, W_ih, W_hh, b_ih, b_hh,
      out0, out1, batch);
}
